// GraphSAGE_53145925321203
// MI455X (gfx1250) — compile-verified, matrix operand order checked by reading, not run on hardware
//
#include <hip/hip_runtime.h>
#include <stdint.h>

#define N_NODES 10000
#define N_EDGES 640000
#define D 128
#define ND (N_NODES * D)

typedef __attribute__((ext_vector_type(16))) __bf16 v16bf;
typedef __attribute__((ext_vector_type(2)))  __bf16 v2bf;
typedef __attribute__((ext_vector_type(8)))  float  v8f;

// ---- f32 -> packed bf16x2 in a u32 -------------------------------------
static __device__ __forceinline__ uint32_t pack_bf(float a, float b) {
#if __has_builtin(__builtin_amdgcn_cvt_pk_bf16_f32)
  v2bf r = __builtin_amdgcn_cvt_pk_bf16_f32(a, b);
  return __builtin_bit_cast(uint32_t, r);
#else
  union { float f; uint32_t u; } ca, cb;
  ca.f = a; cb.f = b;
  uint32_t lo = (ca.u + 0x7FFFu + ((ca.u >> 16) & 1u)) >> 16;
  uint32_t hi = (cb.u + 0x7FFFu + ((cb.u >> 16) & 1u)) >> 16;
  return lo | (hi << 16);
#endif
}

union Frag {
  uint32_t u[8];
  uint4    q[2];
  v16bf    v;
};

// Fragment for 16x32 bf16 WMMA, loaded from row-major bf16 (dword view).
// u32 j holds K = kb+kg+{0,2,4,6,16,18,20,22}[j] (+1 in high half) ->
// dword indices (kb+kg)/2 + {0..3} and + {8..11}: two aligned b128 loads.
static __device__ __forceinline__ v16bf load_frag_bf(
    const uint32_t* __restrict__ rowDw, int kb, int kg) {
  Frag fr;
  const int base = (kb + kg) >> 1;
  fr.q[0] = *(const uint4*)(rowDw + base);
  fr.q[1] = *(const uint4*)(rowDw + base + 8);
  return fr.v;
}

// ---- utility kernels ----------------------------------------------------
__global__ void __launch_bounds__(256) zero_f32(float* __restrict__ p, int n) {
  int i = blockIdx.x * 256 + threadIdx.x;
  if (i < n) p[i] = 0.0f;
}

// f32 -> bf16, 4 elements per thread (n must be divisible by 4)
__global__ void __launch_bounds__(256) cvt_bf16(
    const float* __restrict__ src, uint32_t* __restrict__ dst, int n4) {
  int i = blockIdx.x * 256 + threadIdx.x;
  if (i >= n4) return;
  float4 v = ((const float4*)src)[i];
  uint2 o;
  o.x = pack_bf(v.x, v.y);
  o.y = pack_bf(v.z, v.w);
  ((uint2*)dst)[i] = o;
}

// h_neigh = agg / max(deg,1), converted to bf16 (4 elems/thread)
__global__ void __launch_bounds__(256) cvt_norm_bf16(
    const float* __restrict__ agg, const float* __restrict__ deg,
    uint32_t* __restrict__ dst) {
  int i = blockIdx.x * 256 + threadIdx.x;
  if (i >= ND / 4) return;
  int m = i / (D / 4);
  float inv = 1.0f / fmaxf(deg[m], 1.0f);
  float4 v = ((const float4*)agg)[i];
  uint2 o;
  o.x = pack_bf(v.x * inv, v.y * inv);
  o.y = pack_bf(v.z * inv, v.w * inv);
  ((uint2*)dst)[i] = o;
}

// ---- edge scatter: one edge per 32-lane group, 4 dims per lane ----------
__global__ void __launch_bounds__(256) sage_scatter(
    const float* __restrict__ x, const int* __restrict__ src,
    const int* __restrict__ dst, float* __restrict__ agg,
    float* __restrict__ deg) {
  long long gid = (long long)blockIdx.x * 256 + threadIdx.x;
  int e = (int)(gid >> 5);
  if (e >= N_EDGES) return;
  int c = ((int)gid & 31) * 4;
  int s = src[e];
  int d = dst[e];
  float4 v = *(const float4*)(x + (long long)s * D + c);
  float* a = agg + (long long)d * D + c;
  atomicAdd(a + 0, v.x);
  atomicAdd(a + 1, v.y);
  atomicAdd(a + 2, v.z);
  atomicAdd(a + 3, v.w);
  if ((gid & 31) == 0) atomicAdd(deg + d, 1.0f);
}

// ---- fused GEMM: out = relu( X@Ws^T + Hn@Wn^T + b ) ---------------------
// All operands pre-converted to row-major bf16 (dword pointers).
// Block = 256 = 8 waves; block covers 16 rows, wave w covers cols [16w,16w+16).
__global__ void __launch_bounds__(256) sage_gemm(
    const uint32_t* __restrict__ xbf, const uint32_t* __restrict__ gbf,
    const uint32_t* __restrict__ wsbf, const uint32_t* __restrict__ wnbf,
    const float* __restrict__ bias, float* __restrict__ out) {
  const int lane = threadIdx.x & 31;
  const int wave = threadIdx.x >> 5;
  const int m0 = blockIdx.x * 16;
  const int n0 = wave * 16;
  const int lrow = lane & 15;
  const int kg = (lane >> 4) * 8;
  const int DW = D / 2;  // dwords per row

  const uint32_t* __restrict__ xrow = xbf  + (m0 + lrow) * DW;  // A rows
  const uint32_t* __restrict__ grow = gbf  + (m0 + lrow) * DW;
  const uint32_t* __restrict__ wsrow = wsbf + (n0 + lrow) * DW; // B col = W row
  const uint32_t* __restrict__ wnrow = wnbf + (n0 + lrow) * DW;

  v8f acc = {0.f, 0.f, 0.f, 0.f, 0.f, 0.f, 0.f, 0.f};

#pragma unroll
  for (int kb = 0; kb < D; kb += 32) {
    v16bf a = load_frag_bf(xrow, kb, kg);
    v16bf b = load_frag_bf(wsrow, kb, kg);
    acc = __builtin_amdgcn_wmma_f32_16x16x32_bf16(
        false, a, false, b, (short)0, acc, false, false);
  }
#pragma unroll
  for (int kb = 0; kb < D; kb += 32) {
    v16bf a = load_frag_bf(grow, kb, kg);
    v16bf b = load_frag_bf(wnrow, kb, kg);
    acc = __builtin_amdgcn_wmma_f32_16x16x32_bf16(
        false, a, false, b, (short)0, acc, false, false);
  }

  // C/D layout: VGPR v -> M = v + 8*(lane>=16); N = lane%16
  const int mBase = m0 + ((lane >> 4) * 8);
  const int nCol = n0 + (lane & 15);
  const float bn = bias[nCol];
#pragma unroll
  for (int v = 0; v < 8; ++v) {
    float val = acc[v] + bn;
    out[(long long)(mBase + v) * D + nCol] = fmaxf(val, 0.0f);
  }
}

// ---- driver -------------------------------------------------------------
extern "C" void kernel_launch(void* const* d_in, const int* in_sizes, int n_in,
                              void* d_out, int out_size, void* d_ws, size_t ws_size,
                              hipStream_t stream) {
  (void)in_sizes; (void)n_in; (void)out_size; (void)ws_size;

  const float* x   = (const float*)d_in[0];
  const int*   src = (const int*)d_in[1];
  const int*   dst = (const int*)d_in[2];
  const float* W1s = (const float*)d_in[3];
  const float* W1n = (const float*)d_in[4];
  const float* b1  = (const float*)d_in[5];
  const float* W2s = (const float*)d_in[6];
  const float* W2n = (const float*)d_in[7];
  const float* b2  = (const float*)d_in[8];
  float* out = (float*)d_out;

  // workspace layout (16B-aligned sections):
  float* agg = (float*)d_ws;                    // ND f32
  float* deg = agg + ND;                        // 10000 f32 (padded 10240)
  float* h   = deg + 10240;                     // ND f32 (layer-1 output)
  uint32_t* xbf  = (uint32_t*)(h + ND);         // ND/2 u32 (input, bf16)
  uint32_t* gbf  = xbf + ND / 2;                // ND/2 u32 (neigh mean, bf16)
  uint32_t* wsbf = gbf + ND / 2;                // 8192 u32 (W_self, bf16)
  uint32_t* wnbf = wsbf + 8192;                 // 8192 u32 (W_neigh, bf16)

  dim3 blk(256);
  const int scatterBlocks = (N_EDGES * 32) / 256;  // 80000
  const int gemmBlocks = N_NODES / 16;             // 625
  const int cvtXBlocks = (ND / 4 + 255) / 256;     // 1250
  const int cvtWBlocks = (D * D / 4 + 255) / 256;  // 16

  // ---- layer 1 ----
  zero_f32<<<(ND + 255) / 256, blk, 0, stream>>>(agg, ND);
  zero_f32<<<(N_NODES + 255) / 256, blk, 0, stream>>>(deg, N_NODES);
  sage_scatter<<<scatterBlocks, blk, 0, stream>>>(x, src, dst, agg, deg);
  cvt_bf16<<<cvtXBlocks, blk, 0, stream>>>(x, xbf, ND / 4);
  cvt_bf16<<<cvtWBlocks, blk, 0, stream>>>(W1s, wsbf, D * D / 4);
  cvt_bf16<<<cvtWBlocks, blk, 0, stream>>>(W1n, wnbf, D * D / 4);
  cvt_norm_bf16<<<cvtXBlocks, blk, 0, stream>>>(agg, deg, gbf);
  sage_gemm<<<gemmBlocks, blk, 0, stream>>>(xbf, gbf, wsbf, wnbf, b1, h);

  // ---- layer 2 ----
  zero_f32<<<(ND + 255) / 256, blk, 0, stream>>>(agg, ND);
  zero_f32<<<(N_NODES + 255) / 256, blk, 0, stream>>>(deg, N_NODES);
  sage_scatter<<<scatterBlocks, blk, 0, stream>>>(h, src, dst, agg, deg);
  cvt_bf16<<<cvtXBlocks, blk, 0, stream>>>(h, xbf, ND / 4);
  cvt_bf16<<<cvtWBlocks, blk, 0, stream>>>(W2s, wsbf, D * D / 4);
  cvt_bf16<<<cvtWBlocks, blk, 0, stream>>>(W2n, wnbf, D * D / 4);
  cvt_norm_bf16<<<cvtXBlocks, blk, 0, stream>>>(agg, deg, gbf);
  sage_gemm<<<gemmBlocks, blk, 0, stream>>>(xbf, gbf, wsbf, wnbf, b2, out);
}
